// Pose_GCN_21208548508462
// MI455X (gfx1250) — matrix operand order repaired and checked by reading, not compiled, not run on hardware
//
#include <hip/hip_runtime.h>
#include <hip/hip_fp16.h>

// ---------------------------------------------------------------------------
// Pose-GCN forward for MI455X (gfx1250, wave32, WMMA + Tensor Data Mover).
// Dense layers: v_wmma_f32_16x16x32_f16; operand slabs staged to LDS by TDM.
// ---------------------------------------------------------------------------

#define V_      5
#define NB_     2560
#define KJ      15
#define NBK     (NB_*KJ)          // 38400
#define M1_     (V_*NBK)          // 192000
#define M2_     (NBK)             // 38400
#define SCALAR_ 100.0f
#define KPMAX   288               // largest GEMM K (encode layer)

typedef __attribute__((ext_vector_type(16))) _Float16 v16h;
typedef __attribute__((ext_vector_type(8)))  float    v8f;
typedef _Float16 fh;

enum { F_RELU = 1, F_F32 = 2, F_F16 = 4, F_STATS = 8 };

#ifndef __has_builtin
#define __has_builtin(x) 0
#endif
#if __has_builtin(__builtin_amdgcn_tensor_load_to_lds) && \
    __has_builtin(__builtin_amdgcn_s_wait_tensorcnt)
#define USE_TDM 1
#else
#define USE_TDM 0
#endif

// Normalized adjacency for the 15-joint skeleton: AHAT = D^-1/2 (A+I) D^-1/2.
__constant__ unsigned short c_adj[KJ] = {
  0x0003, 0x4927, 0x000E, 0x001C, 0x0018, 0x0062, 0x00E0, 0x00C0,
  0x0302, 0x0700, 0x0600, 0x1802, 0x3800, 0x3000, 0x4002 };
__constant__ float c_dinv[KJ] = {
  0.70710678f, 0.37796447f, 0.57735027f, 0.57735027f, 0.70710678f,
  0.57735027f, 0.57735027f, 0.70710678f, 0.57735027f, 0.57735027f,
  0.70710678f, 0.57735027f, 0.57735027f, 0.70710678f, 0.70710678f };

#if USE_TDM
typedef unsigned int v4u  __attribute__((ext_vector_type(4)));
typedef int          v8i_ __attribute__((ext_vector_type(8)));
typedef int          v4i_ __attribute__((ext_vector_type(4)));

// Issue one 2D TENSOR_LOAD_TO_LDS: `rows` rows of `rowlen` fp16 elements,
// row stride `stride_e` elements, hardware zero-fill past `tensor_rows`.
// D# layout per CDNA5 ISA ch.8 (group0: count/lds/global/type, group1: dims).
// clang-23 toolchain exposes the 6-operand builtin:
//   (uint32x4 g0, int32x8 g1, int32x4 g2, int32x4 g3, int32x8 pad, i32 cpol)
__device__ __forceinline__ void tdm_load_2d(unsigned lds_off, const void* gptr,
                                            unsigned rowlen, unsigned rows,
                                            unsigned stride_e, unsigned tensor_rows)
{
  const unsigned long long ga = (unsigned long long)(size_t)gptr;
  v4u g0;
  g0.x = 1u;                                            // count=1 (user D#)
  g0.y = lds_off;                                       // lds_addr [63:32]
  g0.z = (unsigned)(ga & 0xFFFFFFFFull);                // global_addr[31:0]
  g0.w = (unsigned)((ga >> 32) & 0x01FFFFFFull)         // global_addr[56:32]
         | (2u << 30);                                  // type = 2 ("image")
  v8i_ g1;
  g1[0] = (1 << 16);                                    // data_size = 2 bytes
  g1[1] = (int)((rowlen & 0xFFFFu) << 16);              // tensor_dim0 lo16 @63:48
  g1[2] = (int)(((rowlen >> 16) & 0xFFFFu)              // tensor_dim0 hi16
         | ((tensor_rows & 0xFFFFu) << 16));            // tensor_dim1 lo16
  g1[3] = (int)(((tensor_rows >> 16) & 0xFFFFu)         // tensor_dim1 hi16
         | ((rowlen & 0xFFFFu) << 16));                 // tile_dim0 = rowlen
  g1[4] = (int)(rows & 0xFFFFu);                        // tile_dim1 (tile_dim2=0)
  g1[5] = (int)stride_e;                                // tensor_dim0_stride[31:0]
  g1[6] = 0;                                            // stride hi / dim1_stride
  g1[7] = 0;
  const v4i_ z4 = {0, 0, 0, 0};                         // groups 2/3 unused (2D)
  const v8i_ z8 = {0, 0, 0, 0, 0, 0, 0, 0};
  __builtin_amdgcn_tensor_load_to_lds(g0, g1, z4, z4, z8, 0);
}
#endif

// ------------------------- WMMA fragment loaders ---------------------------
// A (16x32 f16): lanes 0-15 -> M=0..15 with K {0..7,16..23}; lanes 16-31 hold
// K {8..15,24..31} (ISA 7.12.2).  Two contiguous 16B chunks per lane.
__device__ __forceinline__ v16h load_a_frag(const fh* __restrict__ A, int lda,
                                            int row0, int k0, int lane) {
  const int m  = lane & 15;
  const int hk = (lane >> 4) * 8;
  const fh* p = A + (size_t)(row0 + m) * lda + (k0 + hk);
  v16h a;
#pragma unroll
  for (int e = 0; e < 8; ++e) a[e] = p[e];
  p += 16;
#pragma unroll
  for (int e = 0; e < 8; ++e) a[8 + e] = p[e];
  return a;
}

// B (32x16 f16): lane -> column N=lane&15, K-halfspace by lane>>4.  Weights are
// packed channel-major (Wt[n*Kp + k]) so each lane does two 16B contiguous loads.
__device__ __forceinline__ v16h load_b_frag(const fh* __restrict__ Wt, int ldw,
                                            int col0, int k0, int lane) {
  const int n  = lane & 15;
  const int kb = (lane >> 4) * 8;
  const fh* p = Wt + (size_t)(col0 + n) * ldw + (k0 + kb);
  v16h b;
#pragma unroll
  for (int e = 0; e < 16; ++e) b[e] = p[(e & 7) + ((e >> 3) << 4)];
  return b;
}

// --------------------------- main WMMA GEMM --------------------------------
// C[M,N] = A[M,Kp](f16) * Wt[N,Kp](f16)^T + bias.  8 waves/block, each wave a
// 16x16 tile, block tile 64x32.  The 64xKp A slab and 32xKp B slab are staged
// into LDS with two TENSOR_LOAD_TO_LDS DMAs (wave 0 issues, TENSORcnt wait,
// workgroup barrier), then the k-loop runs entirely out of LDS.
// Optional: ReLU, f32 store, f16 store (ldc16), per-channel sum/sumsq stats.
__global__ __launch_bounds__(256) void k_gemm(
    const fh* __restrict__ A, const fh* __restrict__ Wt,
    const float* __restrict__ bias,
    float* __restrict__ Cf, fh* __restrict__ Ch,
    float* __restrict__ stats,
    int M, int N, int Kp, int ldc16, int flags)
{
  __shared__ float s_sum[32];
  __shared__ float s_sq[32];
#if USE_TDM
  __shared__ fh sA[64 * KPMAX];
  __shared__ fh sB[32 * KPMAX];
#endif
  const int lane = threadIdx.x & 31;
  const int wave = threadIdx.x >> 5;
  const int rloc = (wave & 3) * 16;
  const int cloc = (wave >> 2) * 16;
  const int rowB = blockIdx.x * 64;
  const int colB = blockIdx.y * 32;
  const int row0 = rowB + rloc;
  const int col0 = colB + cloc;
  const bool active = (row0 < M) && (col0 < N);

  if (flags & F_STATS) {
    if (threadIdx.x < 32) { s_sum[threadIdx.x] = 0.f; s_sq[threadIdx.x] = 0.f; }
  }

#if USE_TDM
  if (threadIdx.x < 32) {           // wave 0 drives the TDM (per-wave counter)
    tdm_load_2d((unsigned)(size_t)(void*)sA, A + (size_t)rowB * Kp,
                (unsigned)Kp, 64u, (unsigned)Kp, (unsigned)(M - rowB));
    const int nb = (N - colB) < 32 ? (N - colB) : 32;   // OOB rows zero-filled
    tdm_load_2d((unsigned)(size_t)(void*)sB, Wt + (size_t)colB * Kp,
                (unsigned)Kp, 32u, (unsigned)Kp, (unsigned)nb);
    __builtin_amdgcn_s_wait_tensorcnt(0);
  }
  __syncthreads();
#else
  if (flags & F_STATS) __syncthreads();
#endif

  if (active) {
    v8f acc = {0.f, 0.f, 0.f, 0.f, 0.f, 0.f, 0.f, 0.f};
    for (int k0 = 0; k0 < Kp; k0 += 32) {
#if USE_TDM
      v16h af = load_a_frag(sA, Kp, rloc, k0, lane);
      v16h bf = load_b_frag(sB, Kp, cloc, k0, lane);
#else
      v16h af = load_a_frag(A, Kp, row0, k0, lane);
      v16h bf = load_b_frag(Wt, Kp, col0, k0, lane);
#endif
      acc = __builtin_amdgcn_wmma_f32_16x16x32_f16(
          false, af, false, bf, (short)0, acc, false, false);
    }
    const int   n     = col0 + (lane & 15);
    const float bv    = bias ? bias[n] : 0.f;
    const int   rbase = row0 + (lane >> 4) * 8;   // C layout: VGPR r -> M=r / 8+r
    float lsum = 0.f, lsq = 0.f;
#pragma unroll
    for (int r = 0; r < 8; ++r) {
      float v = acc[r] + bv;
      if (flags & F_RELU) v = fmaxf(v, 0.f);
      const size_t m = (size_t)(rbase + r);
      if (flags & F_F32) Cf[m * (size_t)N + n] = v;
      if (flags & F_F16) Ch[m * (size_t)ldc16 + n] = (fh)v;
      lsum += v; lsq += v * v;
    }
    if (flags & F_STATS) {
      const int nl = (wave >> 2) * 16 + (lane & 15);
      atomicAdd(&s_sum[nl], lsum);
      atomicAdd(&s_sq[nl], lsq);
    }
  }
  if (flags & F_STATS) {
    __syncthreads();
    if (threadIdx.x < 32) {
      const int n = blockIdx.y * 32 + threadIdx.x;
      atomicAdd(&stats[n],       s_sum[threadIdx.x]);
      atomicAdd(&stats[256 + n], s_sq[threadIdx.x]);
    }
  }
}

// ------------------------- graph mix: Y = AHAT @ X -------------------------
__global__ __launch_bounds__(256) void k_ahat(const fh* __restrict__ Xin,
                                              fh* __restrict__ Xout,
                                              int G, int ld)
{
  size_t t = (size_t)blockIdx.x * blockDim.x + threadIdx.x;
  if (t >= (size_t)G * ld) return;
  const int    c    = (int)(t % ld);
  const size_t g    = t / ld;
  const size_t base = g * (size_t)(KJ * ld) + c;
  float x[KJ];
#pragma unroll
  for (int j = 0; j < KJ; ++j)
    x[j] = (float)Xin[base + (size_t)j * ld] * c_dinv[j];
#pragma unroll
  for (int k = 0; k < KJ; ++k) {
    const unsigned m = c_adj[k];
    float a = 0.f;
#pragma unroll
    for (int j = 0; j < KJ; ++j)
      if ((m >> j) & 1) a += x[j];
    Xout[base + (size_t)k * ld] = (fh)(a * c_dinv[k]);
  }
}

// ------------------------ BatchNorm finalize / apply -----------------------
__global__ void k_bnfin(const float* __restrict__ stats,
                        const float* __restrict__ g, const float* __restrict__ b,
                        float* __restrict__ scsh, int N, float invM)
{
  int c = blockIdx.x * blockDim.x + threadIdx.x;
  if (c >= N) return;
  float m   = stats[c] * invM;
  float var = stats[256 + c] * invM - m * m;
  float sc  = g[c] * rsqrtf(var + 1e-5f);
  scsh[c]       = sc;
  scsh[256 + c] = b[c] - m * sc;
}

__global__ __launch_bounds__(256) void k_bnrelu(
    const float* __restrict__ Z, const float* __restrict__ scsh,
    fh* __restrict__ outh, float* __restrict__ outf,
    int M, int N, int ldo)
{
  size_t i = (size_t)blockIdx.x * blockDim.x + threadIdx.x;
  if (i >= (size_t)M * N) return;
  int    c   = (int)(i % N);
  size_t row = i / N;
  float y = fmaxf(fmaf(Z[i], scsh[c], scsh[256 + c]), 0.f);
  outh[row * (size_t)ldo + c] = (fh)y;
  if (outf) outf[i] = y;
}

// ------------------------------ preprocessing ------------------------------
__global__ void k_prep_graph(const float* __restrict__ i3d,
                             float* __restrict__ meanb, int* __restrict__ valid,
                             float* __restrict__ origin, fh* __restrict__ pos4)
{
  int n = blockIdx.x * blockDim.x + threadIdx.x;
  if (n >= NB_) return;
  const float* p = i3d + (size_t)n * KJ * 4;
  int cnt = 0;
  for (int k = 0; k < KJ; ++k) cnt += (p[k * 4 + 3] > 0.f);
  const int vl = (cnt > 3);
  valid[n] = vl;
  const float mx = p[2*4+0] / SCALAR_, my = p[2*4+1] / SCALAR_, mz = p[2*4+2] / SCALAR_;
  meanb[n*3+0] = mx; meanb[n*3+1] = my; meanb[n*3+2] = mz;
  for (int k = 0; k < KJ; ++k) {
    float x = p[k*4+0] / SCALAR_, y = p[k*4+1] / SCALAR_, z = p[k*4+2] / SCALAR_;
    if (vl) { x -= mx; y -= my; z -= mz; }
    const size_t r = (size_t)(n * KJ + k);
    origin[r*3+0] = x; origin[r*3+1] = y; origin[r*3+2] = z;
    fh* q = pos4 + r * 32;
    q[0] = (fh)x; q[1] = (fh)y; q[2] = (fh)z;
    q[3] = (fh)((p[k*4+3] > 0.f) ? 1.f : 0.f);
    for (int c = 4; c < 32; ++c) q[c] = (fh)0.f;
  }
}

__global__ void k_prep_svf(const float* __restrict__ ps,
                           const float* __restrict__ meanb,
                           const int* __restrict__ valid,
                           float* __restrict__ score, float* __restrict__ bak3,
                           fh* __restrict__ svf)
{
  int t = blockIdx.x * blockDim.x + threadIdx.x;
  if (t >= V_ * NB_) return;
  const int n = t % NB_;
  const float* p = ps + (size_t)t * KJ * 7;
  float sc = 0.f;
  for (int k = 0; k < KJ; ++k) sc += p[k*7+6];
  score[t] = sc;
  const int vl = valid[n];
  const float mx = meanb[n*3], my = meanb[n*3+1], mz = meanb[n*3+2];
  for (int k = 0; k < KJ; ++k) {
    float f[7];
    for (int c = 0; c < 7; ++c) f[c] = p[k*7+c];
    if (vl) { f[0] = f[0]/SCALAR_ - mx; f[1] = f[1]/SCALAR_ - my; f[2] = f[2]/SCALAR_ - mz; }
    const size_t r = (size_t)t * KJ + k;
    bak3[r*3+0] = f[0]; bak3[r*3+1] = f[1]; bak3[r*3+2] = f[2];
    fh* q = svf + r * 32;
    for (int c = 0; c < 7;  ++c) q[c] = (fh)f[c];
    for (int c = 7; c < 32; ++c) q[c] = (fh)0.f;
  }
}

// write score into column 256 of the 288-wide encode input (cols 257..287 = 0)
__global__ void k_score(const float* __restrict__ score, fh* __restrict__ buf288)
{
  size_t t = (size_t)blockIdx.x * blockDim.x + threadIdx.x;
  if (t >= (size_t)M1_ * 32) return;
  const size_t row = t >> 5;
  const int    c   = (int)(t & 31);
  const int    g   = (int)(row / KJ);
  buf288[row * 288 + 256 + c] = (fh)((c == 0) ? score[g] : 0.f);
}

// --------------------------- attention cores -------------------------------
// self-attn: per (view, token, head) attend over the 4 other views (q/k/v in
// one fused 768-wide projection buffer, head dim 64, scale 1/8).
__global__ __launch_bounds__(256) void k_attn_self(const fh* __restrict__ qkv,
                                                   fh* __restrict__ O)
{
  int t = blockIdx.x * blockDim.x + threadIdx.x;
  if (t >= V_ * NBK * 4) return;
  const int h = t & 3;
  const int b = (t >> 2) % NBK;
  const int v = (t >> 2) / NBK;
  const fh* q = qkv + (size_t)(v * NBK + b) * 768 + h * 64;
  float s[4]; int rows[4]; int jj = 0;
  for (int j = 0; j < V_; ++j) {
    if (j == v) continue;
    const fh* kp = qkv + (size_t)(j * NBK + b) * 768 + 256 + h * 64;
    float d = 0.f;
    for (int e = 0; e < 64; ++e) d += (float)q[e] * (float)kp[e];
    s[jj] = d * 0.125f; rows[jj] = j; ++jj;
  }
  float mx = fmaxf(fmaxf(s[0], s[1]), fmaxf(s[2], s[3]));
  float w[4], tot = 0.f;
  for (int j = 0; j < 4; ++j) { w[j] = __expf(s[j] - mx); tot += w[j]; }
  const float inv = 1.f / tot;
  float o[64];
  for (int e = 0; e < 64; ++e) o[e] = 0.f;
  for (int j = 0; j < 4; ++j) {
    const fh* vp = qkv + (size_t)(rows[j] * NBK + b) * 768 + 512 + h * 64;
    const float a = w[j] * inv;
    for (int e = 0; e < 64; ++e) o[e] += a * (float)vp[e];
  }
  fh* op = O + (size_t)(v * NBK + b) * 256 + h * 64;
  for (int e = 0; e < 64; ++e) op[e] = (fh)o[e];
}

// cross-attn: per (token, head) attend over all 5 views.
__global__ __launch_bounds__(256) void k_attn_cross(const fh* __restrict__ qc,
                                                    const fh* __restrict__ kv,
                                                    fh* __restrict__ O)
{
  int t = blockIdx.x * blockDim.x + threadIdx.x;
  if (t >= NBK * 4) return;
  const int h = t & 3;
  const int b = t >> 2;
  const fh* q = qc + (size_t)b * 768 + h * 64;
  float s[5];
  for (int j = 0; j < V_; ++j) {
    const fh* kp = kv + (size_t)(j * NBK + b) * 768 + 256 + h * 64;
    float d = 0.f;
    for (int e = 0; e < 64; ++e) d += (float)q[e] * (float)kp[e];
    s[j] = d * 0.125f;
  }
  float mx = s[0];
  for (int j = 1; j < 5; ++j) mx = fmaxf(mx, s[j]);
  float w[5], tot = 0.f;
  for (int j = 0; j < 5; ++j) { w[j] = __expf(s[j] - mx); tot += w[j]; }
  const float inv = 1.f / tot;
  float o[64];
  for (int e = 0; e < 64; ++e) o[e] = 0.f;
  for (int j = 0; j < 5; ++j) {
    const fh* vp = kv + (size_t)(j * NBK + b) * 768 + 512 + h * 64;
    const float a = w[j] * inv;
    for (int e = 0; e < 64; ++e) o[e] += a * (float)vp[e];
  }
  fh* op = O + (size_t)b * 256 + h * 64;
  for (int e = 0; e < 64; ++e) op[e] = (fh)o[e];
}

// residual + LayerNorm over 256 channels; one wave32 per row (8 rows/block).
__global__ __launch_bounds__(256) void k_ln(const float* __restrict__ X,
                                            const float* __restrict__ R,
                                            const float* __restrict__ g,
                                            const float* __restrict__ b,
                                            float* __restrict__ outf,
                                            fh* __restrict__ outh, int Mrows)
{
  const int lane = threadIdx.x & 31;
  const int row  = blockIdx.x * 8 + (threadIdx.x >> 5);
  if (row >= Mrows) return;
  const float* xp = X + (size_t)row * 256;
  const float* rp = R + (size_t)row * 256;
  float xs[8];
  float s = 0.f;
#pragma unroll
  for (int i = 0; i < 8; ++i) { xs[i] = xp[i*32 + lane] + rp[i*32 + lane]; s += xs[i]; }
#pragma unroll
  for (int off = 16; off; off >>= 1) s += __shfl_xor(s, off, 32);
  const float m = s * (1.f / 256.f);
  float vv = 0.f;
#pragma unroll
  for (int i = 0; i < 8; ++i) { const float d = xs[i] - m; vv += d * d; }
#pragma unroll
  for (int off = 16; off; off >>= 1) vv += __shfl_xor(vv, off, 32);
  const float rs = rsqrtf(vv * (1.f / 256.f) + 1e-5f);
#pragma unroll
  for (int i = 0; i < 8; ++i) {
    const int c = i*32 + lane;
    const float y = (xs[i] - m) * rs * g[c] + b[c];
    if (outf) outf[(size_t)row * 256 + c] = y;
    outh[(size_t)row * 256 + c] = (fh)y;
  }
}

// ------------------------------- epilogues ---------------------------------
__global__ void k_refs(const float* __restrict__ bak3, const float* __restrict__ so,
                       const float* __restrict__ meanb, const int* __restrict__ valid,
                       float* __restrict__ out)
{
  int t = blockIdx.x * blockDim.x + threadIdx.x;
  if (t >= M1_ * 3) return;
  const int    c   = t % 3;
  const size_t row = (size_t)(t / 3);
  const int    nb  = (int)(row / KJ) % NB_;
  float r = bak3[row * 3 + c] + so[row * 16 + c];
  if (valid[nb]) r += meanb[nb * 3 + c];
  out[(size_t)M2_ * 3 + row * 3 + c] = r * SCALAR_;   // refined_single region
}

__global__ void k_reff(const float* __restrict__ origin, const float* __restrict__ y,
                       const float* __restrict__ meanb, const int* __restrict__ valid,
                       float* __restrict__ out)
{
  int t = blockIdx.x * blockDim.x + threadIdx.x;
  if (t >= M2_ * 3) return;
  const int    c   = t % 3;
  const size_t row = (size_t)(t / 3);
  const int    nb  = (int)(row / KJ);
  float r = origin[row * 3 + c] + y[row * 16 + c];
  if (valid[nb]) r += meanb[nb * 3 + c];
  out[row * 3 + c] = r * SCALAR_;                     // refined region
}

// --------------------------- weight/bias packing ---------------------------
__global__ void k_packw(const float* __restrict__ W, fh* __restrict__ Wt,
                        int Cout, int Cin, int Np, int Kp)
{
  int t = blockIdx.x * blockDim.x + threadIdx.x;
  if (t >= Np * Kp) return;
  const int k = t % Kp, n = t / Kp;
  Wt[t] = (fh)((n < Cout && k < Cin) ? W[n * Cin + k] : 0.f);
}
__global__ void k_packb(const float* __restrict__ b, float* __restrict__ out,
                        int N, int Np)
{
  int t = blockIdx.x * blockDim.x + threadIdx.x;
  if (t >= Np) return;
  out[t] = (t < N) ? b[t] : 0.f;
}
__global__ void k_zero(float* p, int n)
{
  int t = blockIdx.x * blockDim.x + threadIdx.x;
  if (t < n) p[t] = 0.f;
}

// ------------------------------- host driver -------------------------------
extern "C" void kernel_launch(void* const* d_in, const int* in_sizes, int n_in,
                              void* d_out, int out_size, void* d_ws, size_t ws_size,
                              hipStream_t stream)
{
  (void)in_sizes; (void)n_in; (void)out_size; (void)ws_size;
  const float* ps  = (const float*)d_in[0];
  const float* i3d = (const float*)d_in[1];
  auto P = [&](int i) -> const float* { return (const float*)d_in[i]; };

  char* ws = (char*)d_ws;
  size_t off = 0;
  auto alloc = [&](size_t bytes) -> char* {
    char* p = ws + off;
    off += (bytes + 255) & ~(size_t)255;
    return p;
  };

  fh*    bufP   = (fh*)alloc((size_t)M1_ * 288 * 2);   // activation ping (fp16)
  fh*    tmpA   = (fh*)alloc((size_t)M1_ * 288 * 2);   // AHAT-mixed input (fp16)
  float* Z      = (float*)alloc((size_t)M1_ * 256 * 4);// pre-BN / pre-LN f32
  fh*    enc_h  = (fh*)alloc((size_t)M1_ * 256 * 2);
  float* enc_f  = (float*)alloc((size_t)M1_ * 256 * 4);
  fh*    qkv_h  = (fh*)alloc((size_t)M1_ * 768 * 2);   // fused Q|K|V (reused for cross K|V)
  fh*    O_h    = (fh*)alloc((size_t)M1_ * 256 * 2);
  fh*    nv_h   = (fh*)alloc((size_t)M1_ * 256 * 2);   // new_v (post-LN, fp16)
  float* qf_f   = (float*)alloc((size_t)M2_ * 256 * 4);
  fh*    qf_h   = (fh*)alloc((size_t)M2_ * 256 * 2);
  fh*    qc_h   = (fh*)alloc((size_t)M2_ * 768 * 2);
  fh*    ff_h   = (fh*)alloc((size_t)M2_ * 256 * 2);
  fh*    pos4_h = (fh*)alloc((size_t)M2_ * 32 * 2);
  float* so     = (float*)alloc((size_t)M1_ * 16 * 4);
  float* o2     = (float*)alloc((size_t)M2_ * 16 * 4);
  float* bak3   = (float*)alloc((size_t)M1_ * 3 * 4);
  float* orig   = (float*)alloc((size_t)M2_ * 3 * 4);
  float* meanb  = (float*)alloc((size_t)NB_ * 3 * 4);
  int*   valid  = (int*)alloc((size_t)NB_ * 4);
  float* score  = (float*)alloc((size_t)V_ * NB_ * 4);
  float* stats  = (float*)alloc(512 * 4);
  float* scsh   = (float*)alloc(512 * 4);

  auto packW = [&](const float* w, int cout, int cin, int Np, int Kp) -> fh* {
    fh* p = (fh*)alloc((size_t)Np * Kp * 2);
    int n = Np * Kp;
    k_packw<<<(n + 255) / 256, 256, 0, stream>>>(w, p, cout, cin, Np, Kp);
    return p;
  };
  auto packB = [&](const float* b, int N, int Np) -> float* {
    float* p = (float*)alloc((size_t)Np * 4);
    k_packb<<<(Np + 255) / 256, 256, 0, stream>>>(b, p, N, Np);
    return p;
  };

  // --- pack all weights (fp32 -> fp16, zero-padded to WMMA tiles) ---
  const int sv_dims[4][3] = {{7,32,32},{32,64,32},{64,128,64},{128,256,128}};
  const int i3_dims[4][3] = {{4,32,32},{32,64,32},{64,128,64},{128,256,128}};
  const int su_dims[4][3] = {{256,128,256},{128,128,128},{128,32,128},{32,32,32}};

  fh* wt_sv[4]; float* bp_sv[4];
  fh* wt_i3[4]; float* bp_i3[4];
  fh* wt_su[4]; float* bp_su[4];
  fh* wt_fi[4]; float* bp_fi[4];
  for (int l = 0; l < 4; ++l) {
    wt_sv[l] = packW(P(2  + l*4), sv_dims[l][1], sv_dims[l][0], sv_dims[l][1], sv_dims[l][2]);
    bp_sv[l] = packB(P(3  + l*4), sv_dims[l][1], sv_dims[l][1]);
    wt_i3[l] = packW(P(18 + l*4), i3_dims[l][1], i3_dims[l][0], i3_dims[l][1], i3_dims[l][2]);
    bp_i3[l] = packB(P(19 + l*4), i3_dims[l][1], i3_dims[l][1]);
    wt_su[l] = packW(P(34 + l*4), su_dims[l][1], su_dims[l][0], su_dims[l][1], su_dims[l][2]);
    bp_su[l] = packB(P(35 + l*4), su_dims[l][1], su_dims[l][1]);
    wt_fi[l] = packW(P(50 + l*4), su_dims[l][1], su_dims[l][0], su_dims[l][1], su_dims[l][2]);
    bp_fi[l] = packB(P(51 + l*4), su_dims[l][1], su_dims[l][1]);
  }
  fh*    wt_enc  = packW(P(66), 256, 257, 256, 288);
  float* bp_enc  = packB(P(67), 256, 256);
  fh*    wt_sa_i = packW(P(68), 768, 256, 768, 256);
  float* bp_sa_i = packB(P(69), 768, 768);
  fh*    wt_sa_o = packW(P(70), 256, 256, 256, 256);
  float* bp_sa_o = packB(P(71), 256, 256);
  fh*    wt_ca_i = packW(P(72), 768, 256, 768, 256);
  float* bp_ca_i = packB(P(73), 768, 768);
  fh*    wt_ca_o = packW(P(74), 256, 256, 256, 256);
  float* bp_ca_o = packB(P(75), 256, 256);
  fh*    wt_so   = packW(P(80), 3, 32, 16, 32);
  float* bp_so   = packB(P(81), 3, 16);
  fh*    wt_out  = packW(P(82), 3, 32, 16, 32);
  float* bp_out  = packB(P(83), 3, 16);

  // --- one fused GCN layer: AHAT-mix -> WMMA GEMM (+stats) -> BN+ReLU ---
  auto gcn_layer = [&](fh* X, int ldin, int G, fh* wt, float* bp,
                       const float* gamma, const float* beta,
                       fh* Xout, int ldo, int N, float* f32out) {
    const int M = G * KJ;
    const size_t tot = (size_t)G * ldin;
    k_ahat<<<(unsigned)((tot + 255) / 256), 256, 0, stream>>>(X, tmpA, G, ldin);
    k_zero<<<2, 256, 0, stream>>>(stats, 512);
    dim3 g((unsigned)(M / 64), (unsigned)(N / 32));
    k_gemm<<<g, 256, 0, stream>>>(tmpA, wt, bp, Z, (fh*)nullptr, stats,
                                  M, N, ldin, 0, F_F32 | F_STATS);
    k_bnfin<<<1, 256, 0, stream>>>(stats, gamma, beta, scsh, N, 1.f / (float)M);
    const size_t mn = (size_t)M * N;
    k_bnrelu<<<(unsigned)((mn + 255) / 256), 256, 0, stream>>>(Z, scsh, Xout, f32out, M, N, ldo);
  };

  // ---------------- preprocessing ----------------
  k_prep_graph<<<(NB_ + 255) / 256, 256, 0, stream>>>(i3d, meanb, valid, orig, pos4_h);
  k_prep_svf<<<(V_ * NB_ + 255) / 256, 256, 0, stream>>>(ps, meanb, valid, score, bak3, bufP);

  // ---------------- single_view GCN stack (M1 rows) ----------------
  gcn_layer(bufP, 32,  V_*NB_, wt_sv[0], bp_sv[0], P(4),  P(5),  bufP, 32,  32,  nullptr);
  gcn_layer(bufP, 32,  V_*NB_, wt_sv[1], bp_sv[1], P(8),  P(9),  bufP, 64,  64,  nullptr);
  gcn_layer(bufP, 64,  V_*NB_, wt_sv[2], bp_sv[2], P(12), P(13), bufP, 128, 128, nullptr);
  gcn_layer(bufP, 128, V_*NB_, wt_sv[3], bp_sv[3], P(16), P(17), bufP, 288, 256, nullptr);
  k_score<<<(unsigned)(((size_t)M1_ * 32 + 255) / 256), 256, 0, stream>>>(score, bufP);

  // ---------------- encode: 288 -> 256 + ReLU ----------------
  {
    dim3 g((unsigned)(M1_ / 64), 8);
    k_gemm<<<g, 256, 0, stream>>>(bufP, wt_enc, bp_enc, enc_f, enc_h, nullptr,
                                  M1_, 256, 288, 256, F_RELU | F_F32 | F_F16);
  }

  // ---------------- init3d GCN stack (M2 rows) ----------------
  gcn_layer(pos4_h, 32,  NB_, wt_i3[0], bp_i3[0], P(20), P(21), bufP, 32,  32,  nullptr);
  gcn_layer(bufP,   32,  NB_, wt_i3[1], bp_i3[1], P(24), P(25), bufP, 64,  64,  nullptr);
  gcn_layer(bufP,   64,  NB_, wt_i3[2], bp_i3[2], P(28), P(29), bufP, 128, 128, nullptr);
  gcn_layer(bufP,   128, NB_, wt_i3[3], bp_i3[3], P(32), P(33), qf_h, 256, 256, qf_f);

  // ---------------- self-attention ----------------
  {   // fused Q|K|V projection for all 5 views (K/V shared across views)
    dim3 g((unsigned)(M1_ / 64), 24);
    k_gemm<<<g, 256, 0, stream>>>(enc_h, wt_sa_i, bp_sa_i, nullptr, qkv_h, nullptr,
                                  M1_, 768, 256, 768, F_F16);
  }
  k_attn_self<<<(V_ * NBK * 4 + 255) / 256, 256, 0, stream>>>(qkv_h, O_h);
  {
    dim3 g((unsigned)(M1_ / 64), 8);
    k_gemm<<<g, 256, 0, stream>>>(O_h, wt_sa_o, bp_sa_o, Z, (fh*)nullptr, nullptr,
                                  M1_, 256, 256, 0, F_F32);
  }
  k_ln<<<(M1_ + 7) / 8, 256, 0, stream>>>(Z, enc_f, P(76), P(77), nullptr, nv_h, M1_);

  // ---------------- single_up GCN stack + single_out (all views batched) ----
  gcn_layer(nv_h, 256, V_*NB_, wt_su[0], bp_su[0], P(36), P(37), bufP, 128, 128, nullptr);
  gcn_layer(bufP, 128, V_*NB_, wt_su[1], bp_su[1], P(40), P(41), bufP, 128, 128, nullptr);
  gcn_layer(bufP, 128, V_*NB_, wt_su[2], bp_su[2], P(44), P(45), bufP, 32,  32,  nullptr);
  gcn_layer(bufP, 32,  V_*NB_, wt_su[3], bp_su[3], P(48), P(49), bufP, 32,  32,  nullptr);
  {
    dim3 g((unsigned)(M1_ / 64), 1);
    k_gemm<<<g, 256, 0, stream>>>(bufP, wt_so, bp_so, so, (fh*)nullptr, nullptr,
                                  M1_, 16, 32, 0, F_F32);
  }
  k_refs<<<(M1_ * 3 + 255) / 256, 256, 0, stream>>>(bak3, so, meanb, valid, (float*)d_out);

  // ---------------- cross-attention ----------------
  {
    dim3 g((unsigned)(M2_ / 64), 24);
    k_gemm<<<g, 256, 0, stream>>>(qf_h, wt_ca_i, bp_ca_i, nullptr, qc_h, nullptr,
                                  M2_, 768, 256, 768, F_F16);
  }
  {
    dim3 g((unsigned)(M1_ / 64), 24);
    k_gemm<<<g, 256, 0, stream>>>(nv_h, wt_ca_i, bp_ca_i, nullptr, qkv_h, nullptr,
                                  M1_, 768, 256, 768, F_F16);
  }
  k_attn_cross<<<(NBK * 4 + 255) / 256, 256, 0, stream>>>(qc_h, qkv_h, O_h);
  {
    dim3 g((unsigned)(M2_ / 64), 8);
    k_gemm<<<g, 256, 0, stream>>>(O_h, wt_ca_o, bp_ca_o, Z, (fh*)nullptr, nullptr,
                                  M2_, 256, 256, 0, F_F32);
  }
  k_ln<<<(M2_ + 7) / 8, 256, 0, stream>>>(Z, qf_f, P(78), P(79), nullptr, ff_h, M2_);

  // ---------------- final GCN stack + out ----------------
  gcn_layer(ff_h, 256, NB_, wt_fi[0], bp_fi[0], P(52), P(53), bufP, 128, 128, nullptr);
  gcn_layer(bufP, 128, NB_, wt_fi[1], bp_fi[1], P(56), P(57), bufP, 128, 128, nullptr);
  gcn_layer(bufP, 128, NB_, wt_fi[2], bp_fi[2], P(60), P(61), bufP, 32,  32,  nullptr);
  gcn_layer(bufP, 32,  NB_, wt_fi[3], bp_fi[3], P(64), P(65), bufP, 32,  32,  nullptr);
  {
    dim3 g((unsigned)(M2_ / 64), 1);
    k_gemm<<<g, 256, 0, stream>>>(bufP, wt_out, bp_out, o2, (fh*)nullptr, nullptr,
                                  M2_, 16, 32, 0, F_F32);
  }
  k_reff<<<(M2_ * 3 + 255) / 256, 256, 0, stream>>>(orig, o2, meanb, valid, (float*)d_out);
}
